// EdgeGNN_48962627174425
// MI455X (gfx1250) — matrix operand order repaired and checked by reading, not compiled, not run on hardware
//
#include <hip/hip_runtime.h>
#include <math.h>

typedef __attribute__((ext_vector_type(16))) _Float16 v16h;
typedef __attribute__((ext_vector_type(8)))  _Float16 v8h;
typedef __attribute__((ext_vector_type(8)))  float    v8f;

// ---------------------------------------------------------------------------
// WMMA helpers (CDNA5 gfx1250, wave32): D = A(16x32 f16) * B(32x16 f16) + C(f32)
// ---------------------------------------------------------------------------
__device__ __forceinline__ v8f wmma_f16(v16h a, v16h b, v8f c) {
  // 8 args: (neg_a, A, neg_b, B, c_mod, C, reuse_a, reuse_b)
  return __builtin_amdgcn_wmma_f32_16x16x32_f16(false, a, false, b, (short)0, c,
                                                false, false);
}

// A-fragment: lane holds row (lane%16); K values {b..b+7, b+16..b+23},
// b = (laneHi ? 8 : 0) within a 32-wide K chunk.  Two 16-byte loads.
__device__ __forceinline__ v16h load_a16(const _Float16* __restrict__ row,
                                         int chunk, int laneHi) {
  int base = chunk * 32 + (laneHi ? 8 : 0);
  v8h lo = *(const v8h*)(row + base);
  v8h hi = *(const v8h*)(row + base + 16);
  return __builtin_shufflevector(lo, hi, 0, 1, 2, 3, 4, 5, 6, 7,
                                 8, 9, 10, 11, 12, 13, 14, 15);
}

// B-fragment: lane holds column (lane%16); K values {b..b+7, b+16..b+23},
// b = chunk*32 + (laneHi?8:0).  btRow points at Bt + col*K (Bt = W transposed).
__device__ __forceinline__ v16h load_b16(const _Float16* btRow, int chunk,
                                         int laneHi) {
  int base = chunk * 32 + (laneHi ? 8 : 0);
  v8h lo = *(const v8h*)(btRow + base);
  v8h hi = *(const v8h*)(btRow + base + 16);
  return __builtin_shufflevector(lo, hi, 0, 1, 2, 3, 4, 5, 6, 7,
                                 8, 9, 10, 11, 12, 13, 14, 15);
}

__device__ __forceinline__ float sigf(float x) { return 1.0f / (1.0f + expf(-x)); }

// ---------------------------------------------------------------------------
// Elementwise prologue kernels
// ---------------------------------------------------------------------------
__global__ void k_node_feat(const float* __restrict__ prob,
                            const float* __restrict__ seed,
                            const float* __restrict__ Wn,  // [2][64]
                            const float* __restrict__ bn,
                            float* __restrict__ nf, _Float16* __restrict__ nfh,
                            int N) {
  int i = blockIdx.x * blockDim.x + threadIdx.x;
  if (i >= N * 64) return;
  int n = i >> 6, c = i & 63;
  float v = fmaxf(prob[n] * Wn[c] + seed[n] * Wn[64 + c] + bn[c], 0.0f);
  nf[i] = v;
  nfh[i] = (_Float16)v;
}

__global__ void k_edge_feat(const float* __restrict__ raw,
                            const float* __restrict__ We,  // [1][64]
                            const float* __restrict__ be,
                            _Float16* __restrict__ eh, int E) {
  int i = blockIdx.x * blockDim.x + threadIdx.x;
  if (i >= E * 64) return;
  int e = i >> 6, c = i & 63;
  float v = fmaxf(raw[e] * We[c] + be[c], 0.0f);
  eh[i] = (_Float16)v;
}

// ---------------------------------------------------------------------------
// GEMM: out[e][0:64] = relu( [A1row | A2row] (K=128) @ W[128][64] + bias )
// A1 row gathered through idx1 (or identity when idx1 == nullptr).
// Per wave: 16 rows x 64 cols, 4 K-chunks x 4 col-tiles = 16 WMMAs.
// ---------------------------------------------------------------------------
__global__ void __launch_bounds__(256)
k_gemm_concat(const _Float16* __restrict__ A1, const int* __restrict__ idx1,
              const _Float16* __restrict__ A2, const float* __restrict__ W,
              const float* __restrict__ bias, float* __restrict__ outF,
              _Float16* __restrict__ outH, int nrows) {
  __shared__ __align__(32) _Float16 Bt[64 * 128];  // Bt[c][k] = W[k][c]
  for (int t = threadIdx.x; t < 64 * 128; t += 256) {
    int c = t >> 7, k = t & 127;
    Bt[t] = (_Float16)W[k * 64 + c];
  }
  __syncthreads();

  int lane = threadIdx.x & 31;
  int wave = threadIdx.x >> 5;
  int tile = blockIdx.x * 8 + wave;
  int e0 = tile * 16;
  if (e0 >= nrows) return;
  int r = lane & 15;
  int laneHi = (lane >> 4) & 1;
  int e = e0 + r;

  const _Float16* row1 = A1 + (size_t)(idx1 ? idx1[e] : e) * 64;
  const _Float16* row2 = A2 + (size_t)e * 64;

  v8f acc[4] = {};
#pragma unroll
  for (int kc = 0; kc < 4; ++kc) {
    v16h a = load_a16(kc < 2 ? row1 : row2, kc & 1, laneHi);
#pragma unroll
    for (int ct = 0; ct < 4; ++ct) {
      v16h b = load_b16(Bt + (ct * 16 + r) * 128, kc, laneHi);
      acc[ct] = wmma_f16(a, b, acc[ct]);
    }
  }

  int rbase = laneHi ? 8 : 0;
#pragma unroll
  for (int ct = 0; ct < 4; ++ct) {
    int col = ct * 16 + r;
    float bc = bias[col];
#pragma unroll
    for (int i = 0; i < 8; ++i) {
      size_t o = (size_t)(e0 + rbase + i) * 64 + col;
      float v = fmaxf(acc[ct][i] + bc, 0.0f);
      if (outF) outF[o] = v;
      outH[o] = (_Float16)v;
    }
  }
}

// ---------------------------------------------------------------------------
// GEMM K=64, plain f32 A (identity rows): agg2 = relu(A @ W[64][64] + bias)
// ---------------------------------------------------------------------------
__global__ void __launch_bounds__(256)
k_gemm64_plain(const float* __restrict__ A, const float* __restrict__ W,
               const float* __restrict__ bias, float* __restrict__ outF,
               int nrows) {
  __shared__ __align__(32) _Float16 Bt[64 * 64];
  for (int t = threadIdx.x; t < 64 * 64; t += 256) {
    int c = t >> 6, k = t & 63;
    Bt[t] = (_Float16)W[k * 64 + c];
  }
  __syncthreads();

  int lane = threadIdx.x & 31;
  int tile = blockIdx.x * 8 + (threadIdx.x >> 5);
  int e0 = tile * 16;
  if (e0 >= nrows) return;
  int r = lane & 15, laneHi = (lane >> 4) & 1;
  const float* rowA = A + (size_t)(e0 + r) * 64;

  v8f acc[4] = {};
#pragma unroll
  for (int kc = 0; kc < 2; ++kc) {
    v16h a;
    int base = kc * 32 + (laneHi ? 8 : 0);
#pragma unroll
    for (int j = 0; j < 16; ++j) {
      int k = base + (j < 8 ? j : j + 8);
      a[j] = (_Float16)rowA[k];
    }
#pragma unroll
    for (int ct = 0; ct < 4; ++ct) {
      v16h b = load_b16(Bt + (ct * 16 + r) * 64, kc, laneHi);
      acc[ct] = wmma_f16(a, b, acc[ct]);
    }
  }

  int rbase = laneHi ? 8 : 0;
#pragma unroll
  for (int ct = 0; ct < 4; ++ct) {
    int col = ct * 16 + r;
    float bc = bias[col];
#pragma unroll
    for (int i = 0; i < 8; ++i)
      outF[(size_t)(e0 + rbase + i) * 64 + col] = fmaxf(acc[ct][i] + bc, 0.0f);
  }
}

// ---------------------------------------------------------------------------
// Non-backtracking GEMM: x = relu( (agg[src[e]] - m[rev[e]]) @ W_mp + b )
// ---------------------------------------------------------------------------
__global__ void __launch_bounds__(256)
k_gemm64_nb(const float* __restrict__ agg, const int* __restrict__ src,
            const _Float16* __restrict__ mh, const int* __restrict__ rev,
            const float* __restrict__ W, const float* __restrict__ bias,
            _Float16* __restrict__ outH, int nrows) {
  __shared__ __align__(32) _Float16 Bt[64 * 64];
  for (int t = threadIdx.x; t < 64 * 64; t += 256) {
    int c = t >> 6, k = t & 63;
    Bt[t] = (_Float16)W[k * 64 + c];
  }
  __syncthreads();

  int lane = threadIdx.x & 31;
  int tile = blockIdx.x * 8 + (threadIdx.x >> 5);
  int e0 = tile * 16;
  if (e0 >= nrows) return;
  int r = lane & 15, laneHi = (lane >> 4) & 1;
  int e = e0 + r;
  const float* aggRow = agg + (size_t)src[e] * 64;
  const _Float16* mrevRow = mh + (size_t)rev[e] * 64;

  v8f acc[4] = {};
#pragma unroll
  for (int kc = 0; kc < 2; ++kc) {
    v16h a;
    int base = kc * 32 + (laneHi ? 8 : 0);
#pragma unroll
    for (int j = 0; j < 16; ++j) {
      int k = base + (j < 8 ? j : j + 8);
      a[j] = (_Float16)(aggRow[k] - (float)mrevRow[k]);
    }
#pragma unroll
    for (int ct = 0; ct < 4; ++ct) {
      v16h b = load_b16(Bt + (ct * 16 + r) * 64, kc, laneHi);
      acc[ct] = wmma_f16(a, b, acc[ct]);
    }
  }

  int rbase = laneHi ? 8 : 0;
#pragma unroll
  for (int ct = 0; ct < 4; ++ct) {
    int col = ct * 16 + r;
    float bc = bias[col];
#pragma unroll
    for (int i = 0; i < 8; ++i)
      outH[(size_t)(e0 + rbase + i) * 64 + col] =
          (_Float16)fmaxf(acc[ct][i] + bc, 0.0f);
  }
}

// ---------------------------------------------------------------------------
// Fused GRU: gi = x@W_ih^T + b_ih, gh = h@W_hh^T + b_hh (K=64, 192 cols),
// pointwise gates, in-place h update (f32 state + f16 shadow).
// 48 WMMAs per 16-row tile.
// ---------------------------------------------------------------------------
__global__ void __launch_bounds__(256)
k_gru(const _Float16* __restrict__ xh, const float* __restrict__ Wih,
      const float* __restrict__ Whh, const float* __restrict__ bih,
      const float* __restrict__ bhh, float* __restrict__ hF,
      _Float16* __restrict__ hH, int nrows) {
  __shared__ __align__(32) _Float16 BtI[192 * 64];  // W_ih[c][k] directly
  __shared__ __align__(32) _Float16 BtH[192 * 64];
  for (int t = threadIdx.x; t < 192 * 64; t += 256) {
    BtI[t] = (_Float16)Wih[t];
    BtH[t] = (_Float16)Whh[t];
  }
  __syncthreads();

  int lane = threadIdx.x & 31;
  int tile = blockIdx.x * 8 + (threadIdx.x >> 5);
  int e0 = tile * 16;
  if (e0 >= nrows) return;
  int r = lane & 15, laneHi = (lane >> 4) & 1, rbase = laneHi ? 8 : 0;

  const _Float16* xrow = xh + (size_t)(e0 + r) * 64;
  const _Float16* hrow = hH + (size_t)(e0 + r) * 64;
  v16h ax[2] = {load_a16(xrow, 0, laneHi), load_a16(xrow, 1, laneHi)};
  v16h ah[2] = {load_a16(hrow, 0, laneHi), load_a16(hrow, 1, laneHi)};

#pragma unroll
  for (int ct = 0; ct < 4; ++ct) {
    v8f gir = {}, giz = {}, gin = {}, ghr = {}, ghz = {}, ghn = {};
    int cr = ct * 16 + r;
#pragma unroll
    for (int kc = 0; kc < 2; ++kc) {
      gir = wmma_f16(ax[kc], load_b16(BtI + (size_t)cr * 64, kc, laneHi), gir);
      giz = wmma_f16(ax[kc], load_b16(BtI + (size_t)(64 + cr) * 64, kc, laneHi), giz);
      gin = wmma_f16(ax[kc], load_b16(BtI + (size_t)(128 + cr) * 64, kc, laneHi), gin);
      ghr = wmma_f16(ah[kc], load_b16(BtH + (size_t)cr * 64, kc, laneHi), ghr);
      ghz = wmma_f16(ah[kc], load_b16(BtH + (size_t)(64 + cr) * 64, kc, laneHi), ghz);
      ghn = wmma_f16(ah[kc], load_b16(BtH + (size_t)(128 + cr) * 64, kc, laneHi), ghn);
    }
    float bir = bih[cr], biz = bih[64 + cr], bin = bih[128 + cr];
    float bhr = bhh[cr], bhz = bhh[64 + cr], bhn = bhh[128 + cr];
#pragma unroll
    for (int i = 0; i < 8; ++i) {
      size_t o = (size_t)(e0 + rbase + i) * 64 + cr;
      float hv = hF[o];
      float rg = sigf(gir[i] + bir + ghr[i] + bhr);
      float zg = sigf(giz[i] + biz + ghz[i] + bhz);
      float ng = tanhf(gin[i] + bin + rg * (ghn[i] + bhn));
      float hn = (1.0f - zg) * ng + zg * hv;
      hF[o] = hn;
      hH[o] = (_Float16)hn;
    }
  }
}

// ---------------------------------------------------------------------------
// Scatter-add (segment_sum by tgt) + readout + deltas
// ---------------------------------------------------------------------------
__global__ void k_scatter_f32(const float* __restrict__ vals,
                              const int* __restrict__ tgt,
                              float* __restrict__ agg, int E) {
  int i = blockIdx.x * blockDim.x + threadIdx.x;
  if (i >= E * 64) return;
  int e = i >> 6, c = i & 63;
  atomicAdd(&agg[(size_t)tgt[e] * 64 + c], vals[i]);
}

__global__ void k_scatter_f16(const _Float16* __restrict__ vals,
                              const int* __restrict__ tgt,
                              float* __restrict__ agg, int E) {
  int i = blockIdx.x * blockDim.x + threadIdx.x;
  if (i >= E * 64) return;
  int e = i >> 6, c = i & 63;
  atomicAdd(&agg[(size_t)tgt[e] * 64 + c], (float)vals[i]);
}

__global__ void k_logits(const float* __restrict__ agg2,
                         const float* __restrict__ nf,
                         const float* __restrict__ Wout,  // [128][3]
                         const float* __restrict__ bout,
                         float* __restrict__ marg, int N) {
  int n = blockIdx.x * blockDim.x + threadIdx.x;
  if (n >= N) return;
  float lg[3];
#pragma unroll
  for (int s = 0; s < 3; ++s) {
    float acc = bout[s];
    for (int k = 0; k < 64; ++k) acc += agg2[(size_t)n * 64 + k] * Wout[k * 3 + s];
    for (int k = 0; k < 64; ++k) acc += nf[(size_t)n * 64 + k] * Wout[(64 + k) * 3 + s];
    lg[s] = fmaxf(acc, 0.0f);
  }
  float mx = fmaxf(lg[0], fmaxf(lg[1], lg[2]));
  float se = expf(lg[0] - mx) + expf(lg[1] - mx) + expf(lg[2] - mx);
  float ls = mx + logf(se);
  marg[(size_t)n * 3 + 0] = lg[0] - ls;
  marg[(size_t)n * 3 + 1] = lg[1] - ls;
  marg[(size_t)n * 3 + 2] = lg[2] - ls;
}

__global__ void k_delta(const float* __restrict__ m0, const float* __restrict__ m1,
                        unsigned* __restrict__ dmax, int N) {
  int n = blockIdx.x * blockDim.x + threadIdx.x;
  if (n >= N) return;
  float v = fabsf(expf(m0[(size_t)n * 3 + 2]) - expf(m1[(size_t)n * 3 + 2]));
  atomicMax(dmax, __float_as_uint(v));  // valid ordering: v >= 0
}

__global__ void k_zero4(float* p) {
  if (threadIdx.x < 4) p[threadIdx.x] = 0.0f;
}

// ---------------------------------------------------------------------------
// Host side
// ---------------------------------------------------------------------------
extern "C" void kernel_launch(void* const* d_in, const int* in_sizes, int n_in,
                              void* d_out, int out_size, void* d_ws,
                              size_t ws_size, hipStream_t stream) {
  const int* src = (const int*)d_in[0];
  const int* tgt = (const int*)d_in[1];
  const int* rev = (const int*)d_in[2];
  const float* edge_raw = (const float*)d_in[3];
  const float* prob = (const float*)d_in[4];
  const float* seed = (const float*)d_in[5];
  const float* W_node = (const float*)d_in[6];
  const float* b_node = (const float*)d_in[7];
  const float* W_edge = (const float*)d_in[8];
  const float* b_edge = (const float*)d_in[9];
  const float* W_init = (const float*)d_in[10];
  const float* b_init = (const float*)d_in[11];
  const float* W_aggr = (const float*)d_in[12];
  const float* b_aggr = (const float*)d_in[13];
  const float* W_out = (const float*)d_in[14];
  const float* b_out = (const float*)d_in[15];
  const float* W_me = (const float*)d_in[16];
  const float* b_me = (const float*)d_in[17];
  const float* W_mp = (const float*)d_in[18];
  const float* b_mp = (const float*)d_in[19];
  const float* W_ih = (const float*)d_in[20];
  const float* W_hh = (const float*)d_in[21];
  const float* b_ih = (const float*)d_in[22];
  const float* b_hh = (const float*)d_in[23];

  const int E = in_sizes[0];   // 800000
  const int N = in_sizes[4];   // 50000
  float* out = (float*)d_out;

  // workspace carve (~660 MB)
  char* p = (char*)d_ws;
  auto carve = [&](size_t bytes) {
    void* q = (void*)p;
    p += (bytes + 255) & ~(size_t)255;
    return q;
  };
  float* nodeF = (float*)carve((size_t)N * 64 * 4);
  _Float16* nodeH = (_Float16*)carve((size_t)N * 64 * 2);
  _Float16* edgeH = (_Float16*)carve((size_t)E * 64 * 2);
  float* msgF = (float*)carve((size_t)E * 64 * 4);
  _Float16* msgH = (_Float16*)carve((size_t)E * 64 * 2);
  _Float16* mH = (_Float16*)carve((size_t)E * 64 * 2);
  _Float16* xH = (_Float16*)carve((size_t)E * 64 * 2);
  float* agg = (float*)carve((size_t)N * 64 * 4);
  float* agg2 = (float*)carve((size_t)N * 64 * 4);

  const int eb = ((E + 15) / 16 + 7) / 8;  // wave-tiles of 16 rows, 8 waves/block
  const int nb = ((N + 15) / 16 + 7) / 8;
  const int EC = (E * 64 + 255) / 256;
  const int NC = (N * 64 + 255) / 256;

  k_node_feat<<<NC, 256, 0, stream>>>(prob, seed, W_node, b_node, nodeF, nodeH, N);
  k_edge_feat<<<EC, 256, 0, stream>>>(edge_raw, W_edge, b_edge, edgeH, E);
  k_zero4<<<1, 32, 0, stream>>>(out + (size_t)5 * N * 3);

  // message = relu([node_feat[src] | edge_feat] @ W_init + b_init)
  k_gemm_concat<<<eb, 256, 0, stream>>>(nodeH, src, edgeH, W_init, b_init,
                                        msgF, msgH, E);

  auto nodes_out = [&](int layer) {
    hipMemsetAsync(agg, 0, (size_t)N * 64 * 4, stream);
    k_scatter_f32<<<EC, 256, 0, stream>>>(msgF, tgt, agg, E);
    k_gemm64_plain<<<nb, 256, 0, stream>>>(agg, W_aggr, b_aggr, agg2, N);
    k_logits<<<(N + 255) / 256, 256, 0, stream>>>(
        agg2, nodeF, W_out, b_out, out + (size_t)layer * N * 3, N);
  };

  nodes_out(0);

  for (int l = 0; l < 4; ++l) {
    // m = relu([message | edge_feat] @ W_me + b_me)
    k_gemm_concat<<<eb, 256, 0, stream>>>(msgH, nullptr, edgeH, W_me, b_me,
                                          nullptr, mH, E);
    // node_agg = segment_sum(m, tgt)
    hipMemsetAsync(agg, 0, (size_t)N * 64 * 4, stream);
    k_scatter_f16<<<EC, 256, 0, stream>>>(mH, tgt, agg, E);
    // x = relu((node_agg[src] - m[rev]) @ W_mp + b_mp)
    k_gemm64_nb<<<eb, 256, 0, stream>>>(agg, src, mH, rev, W_mp, b_mp, xH, E);
    // message = GRU(x, message)   (in place)
    k_gru<<<eb, 256, 0, stream>>>(xH, W_ih, W_hh, b_ih, b_hh, msgF, msgH, E);

    nodes_out(l + 1);
    k_delta<<<(N + 255) / 256, 256, 0, stream>>>(
        out + (size_t)l * N * 3, out + (size_t)(l + 1) * N * 3,
        (unsigned*)(out + (size_t)5 * N * 3 + l), N);
  }
}
